// TestModel_65274912964833
// MI455X (gfx1250) — matrix operand order repaired and checked by reading, not compiled, not run on hardware
//
#include <hip/hip_runtime.h>
#include <hip/hip_bf16.h>
#include <math.h>

typedef __attribute__((ext_vector_type(16))) __bf16 v16bf;
typedef __attribute__((ext_vector_type(8)))  float  v8f;

static constexpr int GN_BS   = 64;
static constexpr int GN_G    = 1000;
static constexpr int GN_EPG  = 32;
static constexpr int GN_HID  = 256;
static constexpr int GN_NG   = 768;   // 3*HID
static constexpr int GN_KTOT = 288;   // HID + EPG (layout of Wc)
static constexpr int GN_ROWS = GN_BS * GN_G;          // 64000 gi rows
static constexpr int HBF_STR = 520;   // ushorts per row = 260 dwords (== 4 mod 64 -> conflict-free b128)

__device__ __forceinline__ unsigned short f2bf(float f) {
  unsigned int u = __float_as_uint(f);
  u += 0x7FFFu + ((u >> 16) & 1u);          // round-to-nearest-even
  return (unsigned short)(u >> 16);
}
__device__ __forceinline__ float bf2f(unsigned short h) {
  return __uint_as_float(((unsigned int)h) << 16);
}

// ---------------- Graph layer 1: project x (16 ch) to rel/root (2 ch), zero agg ----------------
__global__ void k_node1(const float* __restrict__ x, const float* __restrict__ W1_rel,
                        const float* __restrict__ W1_root,
                        float* __restrict__ p1, float* __restrict__ r1,
                        float* __restrict__ agg1, int N) {
  __shared__ float wrel[32], wroot[32];
  int tid = threadIdx.x;
  if (tid < 32) { wrel[tid] = W1_rel[tid]; wroot[tid] = W1_root[tid]; }
  __syncthreads();
  int i = blockIdx.x * blockDim.x + tid;
  if (i >= N) return;
  const float* xi = x + (size_t)i * 16;
  float xv[16];
#pragma unroll
  for (int k = 0; k < 16; ++k) xv[k] = xi[k];
  float pa = 0.f, pb = 0.f, qa = 0.f, qb = 0.f;
#pragma unroll
  for (int k = 0; k < 16; ++k) {
    pa += wrel[k] * xv[k];      pb += wrel[16 + k] * xv[k];
    qa += wroot[k] * xv[k];     qb += wroot[16 + k] * xv[k];
  }
  size_t o = (size_t)i * 2;
  p1[o] = pa; p1[o + 1] = pb;
  r1[o] = qa; r1[o + 1] = qb;
  agg1[o] = 0.f; agg1[o + 1] = 0.f;
}

// ---------------- Edge scatter, 2-channel ----------------
__global__ void k_edge1(const int* __restrict__ src, const int* __restrict__ dst,
                        const float* __restrict__ ew, const float* __restrict__ p1,
                        float* __restrict__ agg1, int E) {
  int e = blockIdx.x * blockDim.x + threadIdx.x;
  if (e >= E) return;
  size_t s = (size_t)src[e] * 2, d = (size_t)dst[e] * 2;
  float w = ew[e];
  unsafeAtomicAdd(&agg1[d],     w * p1[s]);
  unsafeAtomicAdd(&agg1[d + 1], w * p1[s + 1]);
}

// ---------------- relu1 + projection to layer-2 (1 ch), zero agg2 ----------------
__global__ void k_node2(const float* __restrict__ agg1, const float* __restrict__ r1,
                        const float* __restrict__ b1, const float* __restrict__ W2_rel,
                        const float* __restrict__ W2_root,
                        float* __restrict__ p2, float* __restrict__ r2,
                        float* __restrict__ agg2, int N) {
  int i = blockIdx.x * blockDim.x + threadIdx.x;
  if (i >= N) return;
  size_t o = (size_t)i * 2;
  float h0 = fmaxf(agg1[o]     + b1[0] + r1[o],     0.f);
  float h1 = fmaxf(agg1[o + 1] + b1[1] + r1[o + 1], 0.f);
  p2[i] = W2_rel[0]  * h0 + W2_rel[1]  * h1;
  r2[i] = W2_root[0] * h0 + W2_root[1] * h1;
  agg2[i] = 0.f;
}

// ---------------- Edge scatter, 1-channel ----------------
__global__ void k_edge2(const int* __restrict__ src, const int* __restrict__ dst,
                        const float* __restrict__ ew, const float* __restrict__ p2,
                        float* __restrict__ agg2, int E) {
  int e = blockIdx.x * blockDim.x + threadIdx.x;
  if (e >= E) return;
  unsafeAtomicAdd(&agg2[dst[e]], ew[e] * p2[src[e]]);
}

// ---------------- relu2 -> sequence ----------------
__global__ void k_node3(const float* __restrict__ agg2, const float* __restrict__ r2,
                        const float* __restrict__ b2, float* __restrict__ h2, int N) {
  int i = blockIdx.x * blockDim.x + threadIdx.x;
  if (i >= N) return;
  h2[i] = fmaxf(agg2[i] + b2[0] + r2[i], 0.f);
}

// ---------------- bf16 convert of [W_hh | W_ih] -> Wc[768][288] ----------------
__global__ void k_wconv(const float* __restrict__ Whh, const float* __restrict__ Wih,
                        unsigned short* __restrict__ Wc) {
  int idx = blockIdx.x * blockDim.x + threadIdx.x;
  if (idx >= GN_NG * GN_KTOT) return;
  int n = idx / GN_KTOT, k = idx - n * GN_KTOT;
  float v = (k < GN_HID) ? Whh[(size_t)n * GN_HID + k] : Wih[(size_t)n * GN_EPG + (k - GN_HID)];
  Wc[idx] = f2bf(v);
}

// ---------------- gi = seq(64000x32) @ W_ih^T + b_ih  ->  bf16 [64000][768] ----------------
// One wave per 16-row tile; loops over 48 column tiles; 1 WMMA per tile (K=32).
__global__ void __launch_bounds__(256)
k_gi(const float* __restrict__ seq, const unsigned short* __restrict__ Wc,
     const float* __restrict__ bih, unsigned short* __restrict__ gi) {
  const int wv    = (blockIdx.x * blockDim.x + threadIdx.x) >> 5;  // 0..3999
  const int lane  = threadIdx.x & 31;
  const int lhalf = lane >> 4;
  const int l16   = lane & 15;
  const int row0  = wv * 16;

  // A fragment (16x32): lane<16 -> row=row0+l16, K=[0..15]; lane>=16 -> K=[16..31]
  const float* ap = seq + (size_t)(row0 + l16) * GN_EPG + lhalf * 16;
  union { v16bf v; unsigned short s[16]; } ua;
#pragma unroll
  for (int k = 0; k < 8; ++k) {
    ua.s[k]     = f2bf(ap[k]);
    ua.s[k + 8] = f2bf(ap[k + 8]);
  }

  for (int nt = 0; nt < GN_NG / 16; ++nt) {
    const int col = nt * 16 + l16;
    const unsigned short* bp = Wc + (size_t)col * GN_KTOT + GN_HID + lhalf * 16;
    union { v16bf v; uint4 q[2]; } ub;
    ub.q[0] = *(const uint4*)(bp);
    ub.q[1] = *(const uint4*)(bp + 8);
    v8f acc = {0.f,0.f,0.f,0.f,0.f,0.f,0.f,0.f};
    acc = __builtin_amdgcn_wmma_f32_16x16x32_bf16(false, ua.v, false, ub.v, (short)0, acc, false, false);
    const float bb = bih[col];
#pragma unroll
    for (int v = 0; v < 8; ++v)
      gi[(size_t)(row0 + v + lhalf * 8) * GN_NG + col] = f2bf(acc[v] + bb);
  }
}

// ---------------- Persistent single-WGP GRU: 24 waves, W_hh in registers, h in LDS ----------------
__global__ void __launch_bounds__(768, 1)
k_gru(const unsigned short* __restrict__ gi, const unsigned short* __restrict__ Wc,
      const float* __restrict__ bhh, float* __restrict__ hOut) {
  __shared__ unsigned short hbf[GN_BS * HBF_STR];   // h state, bf16 (A-frag layout friendly)
  __shared__ float          hx [GN_BS * GN_HID];    // h state, fp32
  __shared__ unsigned short comb[GN_BS * GN_NG];    // h @ W_hh^T staging, bf16
  __shared__ float          bhhS[GN_NG];

  const int tid = threadIdx.x;
  const int wave = tid >> 5;
  const int lane = tid & 31;
  const int lhalf = lane >> 4;     // 0 | 1
  const int l16 = lane & 15;

  for (int i = tid; i < GN_BS * GN_HID; i += 768) hx[i] = 0.f;
  for (int i = tid; i < GN_BS * HBF_STR; i += 768) hbf[i] = 0;
  for (int i = tid; i < GN_NG; i += 768) bhhS[i] = bhh[i];

  // Persistent B fragments: wave owns strips {2w, 2w+1}, 8 K-frags (K=256) => 128 VGPRs.
  // B layout (32x16 bf16): lane<16 -> col N=l16, K = kbase..kbase+15; lane>=16 -> K += 16.
  v16bf Bf[2][8];
  const int strip0 = wave * 2;
#pragma unroll
  for (int s = 0; s < 2; ++s) {
    int n = (strip0 + s) * 16 + l16;
#pragma unroll
    for (int kf = 0; kf < 8; ++kf) {
      const unsigned short* p = Wc + (size_t)n * GN_KTOT + kf * 32 + lhalf * 16;
      union { v16bf v; uint4 q[2]; } u;
      u.q[0] = *(const uint4*)(p);
      u.q[1] = *(const uint4*)(p + 8);
      Bf[s][kf] = u.v;
    }
  }
  __syncthreads();

  const int nb0 = strip0 * 16, nb1 = nb0 + 16;

  for (int t = 0; t < GN_G; ++t) {
    // h (64 x 256) @ W_hh^T  via bf16 WMMA, f32 accumulate
#pragma unroll 1
    for (int mt = 0; mt < 4; ++mt) {
      const unsigned short* pa0 = hbf + (mt * 16 + l16) * HBF_STR + lhalf * 16;
      unsigned short* cb = &comb[(mt * 16 + lhalf * 8) * GN_NG + l16];

      v8f a0c = {0.f,0.f,0.f,0.f,0.f,0.f,0.f,0.f};
      v8f a1c = {0.f,0.f,0.f,0.f,0.f,0.f,0.f,0.f};
#pragma unroll
      for (int kf = 0; kf < 8; ++kf) {
        union { v16bf v; uint4 q[2]; } ua;
        ua.q[0] = *(const uint4*)(pa0 + kf * 32);
        ua.q[1] = *(const uint4*)(pa0 + kf * 32 + 8);
        a0c = __builtin_amdgcn_wmma_f32_16x16x32_bf16(false, ua.v, false, Bf[0][kf], (short)0, a0c, false, false);
        a1c = __builtin_amdgcn_wmma_f32_16x16x32_bf16(false, ua.v, false, Bf[1][kf], (short)0, a1c, false, false);
      }
#pragma unroll
      for (int v = 0; v < 8; ++v) {
        cb[v * GN_NG + nb0] = f2bf(a0c[v]);
        cb[v * GN_NG + nb1] = f2bf(a1c[v]);
      }
    }
    __syncthreads();

    // gate math (fp32): gi from L2-resident global, comb/h from LDS
    for (int u = tid; u < GN_BS * GN_HID; u += 768) {
      int b = u >> 8, j = u & 255;
      const unsigned short* gp = gi + ((size_t)b * GN_G + t) * GN_NG;
      float gr  = bf2f(comb[b * GN_NG + j])              + bf2f(gp[j])            + bhhS[j];
      float gz  = bf2f(comb[b * GN_NG + GN_HID + j])     + bf2f(gp[GN_HID + j])   + bhhS[GN_HID + j];
      float hn  = bf2f(comb[b * GN_NG + 2*GN_HID + j])   + bhhS[2*GN_HID + j];
      float in2 = bf2f(gp[2*GN_HID + j]);                 // b_ih already folded into gi
      float r = 1.f / (1.f + __expf(-gr));
      float z = 1.f / (1.f + __expf(-gz));
      float n = tanhf(in2 + r * hn);
      float hnew = (1.f - z) * n + z * hx[u];
      hx[u] = hnew;
      hbf[b * HBF_STR + j] = f2bf(hnew);
      if (t + 1 < GN_G) {
        __builtin_prefetch(gp + GN_NG + j, 0, 0);
        __builtin_prefetch(gp + GN_NG + GN_HID + j, 0, 0);
        __builtin_prefetch(gp + GN_NG + 2*GN_HID + j, 0, 0);
      }
    }
    __syncthreads();
  }

  for (int i = tid; i < GN_BS * GN_HID; i += 768) hOut[i] = hx[i];
}

// ---------------- lin1 (256->1024) + softmax, one block per batch row ----------------
__global__ void __launch_bounds__(256)
k_lin_softmax(const float* __restrict__ h, const float* __restrict__ W,
              const float* __restrict__ bias, float* __restrict__ out) {
  __shared__ float hrow[256];
  __shared__ float ex[1024];
  __shared__ float red[256];
  int b = blockIdx.x, tid = threadIdx.x;
  hrow[tid] = h[b * 256 + tid];
  __syncthreads();
  float lv[4];
#pragma unroll
  for (int q = 0; q < 4; ++q) {
    int o = tid + q * 256;
    const float* w = W + (size_t)o * 256;
    float acc = 0.f;
    for (int k = 0; k < 256; k += 4) {
      float4 wv = *(const float4*)(w + k);
      acc += wv.x * hrow[k] + wv.y * hrow[k + 1] + wv.z * hrow[k + 2] + wv.w * hrow[k + 3];
    }
    lv[q] = acc + bias[o];
  }
  float m = fmaxf(fmaxf(lv[0], lv[1]), fmaxf(lv[2], lv[3]));
  red[tid] = m; __syncthreads();
  for (int s = 128; s > 0; s >>= 1) { if (tid < s) red[tid] = fmaxf(red[tid], red[tid + s]); __syncthreads(); }
  float M = red[0]; __syncthreads();
  float ssum = 0.f;
#pragma unroll
  for (int q = 0; q < 4; ++q) { float e = __expf(lv[q] - M); ex[tid + q * 256] = e; ssum += e; }
  red[tid] = ssum; __syncthreads();
  for (int s = 128; s > 0; s >>= 1) { if (tid < s) red[tid] += red[tid + s]; __syncthreads(); }
  float inv = 1.f / red[0];
#pragma unroll
  for (int q = 0; q < 4; ++q) out[(size_t)b * 1024 + tid + q * 256] = ex[tid + q * 256] * inv;
}

extern "C" void kernel_launch(void* const* d_in, const int* in_sizes, int n_in,
                              void* d_out, int out_size, void* d_ws, size_t ws_size,
                              hipStream_t stream) {
  const float* x       = (const float*)d_in[0];
  const int*   eidx    = (const int*)  d_in[1];
  const float* ew      = (const float*)d_in[2];
  const float* W1_rel  = (const float*)d_in[3];
  const float* b1      = (const float*)d_in[4];
  const float* W1_root = (const float*)d_in[5];
  const float* W2_rel  = (const float*)d_in[6];
  const float* b2      = (const float*)d_in[7];
  const float* W2_root = (const float*)d_in[8];
  const float* Wih     = (const float*)d_in[9];
  const float* Whh     = (const float*)d_in[10];
  const float* bih     = (const float*)d_in[11];
  const float* bhh     = (const float*)d_in[12];
  const float* Wlin    = (const float*)d_in[13];
  const float* blin    = (const float*)d_in[14];

  const int N = in_sizes[0] / 16;
  const int E = in_sizes[2];
  const int* src = eidx;
  const int* dst = eidx + E;

  char* ws = (char*)d_ws;
  size_t off = 0;
  auto alloc = [&](size_t bytes) -> void* {
    void* p = ws + off; off += (bytes + 255) & ~(size_t)255; return p;
  };
  float* p1   = (float*)alloc((size_t)N * 2 * sizeof(float));
  float* r1   = (float*)alloc((size_t)N * 2 * sizeof(float));
  float* agg1 = (float*)alloc((size_t)N * 2 * sizeof(float));
  float* p2   = (float*)alloc((size_t)N * sizeof(float));
  float* r2   = (float*)alloc((size_t)N * sizeof(float));
  float* agg2 = (float*)alloc((size_t)N * sizeof(float));
  float* h2   = (float*)alloc((size_t)N * sizeof(float));
  unsigned short* Wc = (unsigned short*)alloc((size_t)GN_NG * GN_KTOT * 2);
  unsigned short* gi = (unsigned short*)alloc((size_t)GN_ROWS * GN_NG * 2);
  float* hFin = (float*)alloc(GN_BS * GN_HID * sizeof(float));
  (void)ws_size; (void)n_in; (void)out_size;

  const int nbk = (N + 255) / 256;
  const int ebk = (E + 255) / 256;

  k_node1<<<nbk, 256, 0, stream>>>(x, W1_rel, W1_root, p1, r1, agg1, N);
  k_edge1<<<ebk, 256, 0, stream>>>(src, dst, ew, p1, agg1, E);
  k_node2<<<nbk, 256, 0, stream>>>(agg1, r1, b1, W2_rel, W2_root, p2, r2, agg2, N);
  k_edge2<<<ebk, 256, 0, stream>>>(src, dst, ew, p2, agg2, E);
  k_node3<<<nbk, 256, 0, stream>>>(agg2, r2, b2, h2, N);
  k_wconv<<<(GN_NG * GN_KTOT + 255) / 256, 256, 0, stream>>>(Whh, Wih, Wc);
  k_gi<<<(GN_ROWS / 16) / 8, 256, 0, stream>>>(h2, Wc, bih, gi);   // 4000 waves
  k_gru<<<1, 768, 0, stream>>>(gi, Wc, bhh, hFin);
  k_lin_softmax<<<GN_BS, 256, 0, stream>>>(hFin, Wlin, blin, (float*)d_out);
}
